// TokenBook3D_32109175505224
// MI455X (gfx1250) — matrix operand order repaired and checked by reading, not compiled, not run on hardware
//
#include <hip/hip_runtime.h>
#include <hip/hip_bf16.h>
#include <math.h>

typedef _Float16 f16;
typedef __attribute__((ext_vector_type(16))) _Float16 v16h;
typedef __attribute__((ext_vector_type(8)))  _Float16 v8h;
typedef __attribute__((ext_vector_type(8)))  float    v8f;

union V16 { v16h v; v8h h[2]; };

#define MT   32            // tokens per block
#define PH   264           // LDS half pitch (pad 8 -> bank rotate)
#define PF   264           // LDS float pitch
#define NTOK 131072        // D*H*W
#define CCH  256
#define TT   1024
#define HID  256

// ---------------- device helpers ----------------

__device__ inline v8f wmma_f16(v16h a, v16h b, v8f c) {
  return __builtin_amdgcn_wmma_f32_16x16x32_f16(false, a, false, b, (short)0, c,
                                                false, false);
}

// A fragment (16x32 f16, M x K) from LDS row-major [m][k], pitch PH.
// ISA layout: lane l: m = l&15; elems 0..7 -> k = kt*32 + 8*(l>>4) + [0..7];
//             elems 8..15 -> +16.  Two 16B ds_load_b128 per lane.
__device__ inline v16h load_a(const f16* lds, int mt, int kt, int ln, int hi) {
  const f16* p = lds + (mt * 16 + ln) * PH + kt * 32 + hi * 8;
  V16 r;
  r.h[0] = *(const v8h*)p;
  r.h[1] = *(const v8h*)(p + 16);
  return r.v;
}

// B fragment (32x16 f16, K x N) pre-packed in ws: tile (kt*NT+nt), 512 halfs,
// element offset = lane*16 + e  ->  per-lane contiguous 32B (2x global b128).
__device__ inline v16h load_b(const f16* g, int NT, int kt, int nt, int lane) {
  const f16* p = g + (size_t)(kt * NT + nt) * 512 + lane * 16;
  V16 r;
  r.h[0] = *(const v8h*)p;
  r.h[1] = *(const v8h*)(p + 8);
  return r.v;
}

// ---------------- prep kernels ----------------

// inverse L2 norm of each codebook row
__global__ __launch_bounds__(256) void norm_kernel(const float* __restrict__ book,
                                                   float* __restrict__ inv) {
  __shared__ float red[256];
  const int r = blockIdx.x;
  const float v = book[(size_t)r * CCH + threadIdx.x];
  red[threadIdx.x] = v * v;
  __syncthreads();
  for (int s = 128; s > 0; s >>= 1) {
    if ((int)threadIdx.x < s) red[threadIdx.x] += red[threadIdx.x + s];
    __syncthreads();
  }
  if (threadIdx.x == 0) inv[r] = 1.0f / fmaxf(sqrtf(red[0]), 1e-12f);
}

// Pack a [K,N] (or transposed-source [N,K]) f32 matrix into B-fragment layout.
// B layout: lanes 0-15: n = lane, k = kt*32 + [0..7, 16..23] ; lanes 16-31: n = lane-16,
// k = kt*32 + [8..15, 24..31]  (e = 0..15).
__global__ __launch_bounds__(256) void pack_kernel(f16* __restrict__ dst,
                                                   const float* __restrict__ src,
                                                   int K, int N, int srcNMajor,
                                                   const float* __restrict__ scale) {
  const int i = blockIdx.x * 256 + threadIdx.x;
  if (i >= K * N) return;
  const int NT = N >> 4;
  const int tile = i >> 9;
  const int r = i & 511;
  const int l = r >> 4;
  const int e = r & 15;
  const int kt = tile / NT;
  const int nt = tile - kt * NT;
  const int k = kt * 32 + ((l >> 4) << 3) + (e & 7) + ((e >> 3) << 4);
  const int n = nt * 16 + (l & 15);
  float v = srcNMajor ? src[(size_t)n * K + k] : src[(size_t)k * N + n];
  if (scale) v *= scale[n];
  dst[i] = (f16)v;
}

// ---------------- fused main kernel ----------------

__global__ __launch_bounds__(256) void tokenbook_fused(
    const float* __restrict__ x, const float* __restrict__ conv_b,
    const float* __restrict__ mlp_b1, const float* __restrict__ mlp_b2,
    const f16* __restrict__ convB, const f16* __restrict__ w1B,
    const f16* __restrict__ w2B, const f16* __restrict__ protosB,
    float* __restrict__ out) {
  const int tid = threadIdx.x;
  const int lane = tid & 31;
  const int w = tid >> 5;     // wave 0..7
  const int hi = lane >> 4;   // lane-half
  const int ln = lane & 15;

  const int g0 = blockIdx.x * MT;      // first global token of tile
  const int b = g0 >> 17;              // / NTOK
  const int n0 = g0 & (NTOK - 1);
  const float* xb = x + (size_t)b * CCH * NTOK;

  __shared__ f16 xh[MT * PH];
  __shared__ f16 tokh[MT * PH];
  __shared__ f16 hh[MT * PH];
  __shared__ float facc[MT * PF];
  __shared__ float parts[8][MT][3];

  // ---- stage x tile [32 tok][256 ch] as f16 (coalesced 128B per channel) ----
  for (int it = 0; it < (MT * CCH) / 256; ++it) {
    const int idx = it * 256 + tid;
    const int c = idx >> 5;
    const int tk = idx & 31;
    xh[tk * PH + c] = (f16)xb[(size_t)c * NTOK + n0 + tk];
  }
  __syncthreads();

  // ---- conv (tokens = x @ conv_w^T): [32x256] x [256x256], WMMA ----
  // Wave owns n-tiles {w, w+8}; B fragment shared across both m-tiles.
  for (int nt = w; nt < 16; nt += 8) {
    v8f a0 = {}, a1 = {};
    for (int kt = 0; kt < 8; ++kt) {
      v16h bb = load_b(convB, 16, kt, nt, lane);
      a0 = wmma_f16(load_a(xh, 0, kt, ln, hi), bb, a0);
      a1 = wmma_f16(load_a(xh, 1, kt, ln, hi), bb, a1);
    }
    float* cp = facc + nt * 16 + ln;
#pragma unroll
    for (int v = 0; v < 8; ++v) {
      cp[(v + 8 * hi) * PF] = a0[v];
      cp[(16 + v + 8 * hi) * PF] = a1[v];
    }
  }
  __syncthreads();

  // ---- bias + L2 normalize each token row (wave32 xor reduction) ----
  for (int m = w * 4; m < w * 4 + 4; ++m) {
    float s = 0.0f;
    float vals[8];
#pragma unroll
    for (int j = 0; j < 8; ++j) {
      const float vv = facc[m * PF + j * 32 + lane] + conv_b[j * 32 + lane];
      vals[j] = vv;
      s += vv * vv;
    }
    for (int off = 16; off >= 1; off >>= 1) s += __shfl_xor(s, off, 32);
    const float inv = 1.0f / fmaxf(sqrtf(s), 1e-12f);
#pragma unroll
    for (int j = 0; j < 8; ++j)
      tokh[m * PH + j * 32 + lane] = (f16)(vals[j] * inv);
  }
  __syncthreads();

  // ---- MLP layer 1: h = gelu(tokens @ w1 + b1) ----
  for (int nt = w; nt < 16; nt += 8) {
    v8f a0 = {}, a1 = {};
    for (int kt = 0; kt < 8; ++kt) {
      v16h bb = load_b(w1B, 16, kt, nt, lane);
      a0 = wmma_f16(load_a(tokh, 0, kt, ln, hi), bb, a0);
      a1 = wmma_f16(load_a(tokh, 1, kt, ln, hi), bb, a1);
    }
    float* cp = facc + nt * 16 + ln;
#pragma unroll
    for (int v = 0; v < 8; ++v) {
      cp[(v + 8 * hi) * PF] = a0[v];
      cp[(16 + v + 8 * hi) * PF] = a1[v];
    }
  }
  __syncthreads();
  for (int i = tid; i < MT * HID; i += 256) {
    const int m = i >> 8, c = i & 255;
    const float vv = facc[m * PF + c] + mlp_b1[c];
    const float gl = 0.5f * vv * (1.0f + erff(vv * 0.70710678118f));
    hh[m * PH + c] = (f16)gl;
  }
  __syncthreads();

  // ---- streamed T loop: logits & sims tiles + online softmax-weighted sum ----
  // Each wave owns 8 of the 64 T-tiles; per-lane accumulators cover tokens
  // m = mt*16 + v + 8*hi (replicated across its 16-lane group).
  float rmax[2][8], rden[2][8], rnum[2][8];
#pragma unroll
  for (int mt = 0; mt < 2; ++mt)
#pragma unroll
    for (int v = 0; v < 8; ++v) {
      rmax[mt][v] = -1e30f;
      rden[mt][v] = 0.0f;
      rnum[mt][v] = 0.0f;
    }

  for (int j = 0; j < 8; ++j) {
    const int tt = w + j * 8;  // T tile 0..63
    v8f lg[2] = {v8f{}, v8f{}};
    v8f sm[2] = {v8f{}, v8f{}};
    for (int kt = 0; kt < 8; ++kt) {
      // B fragments loaded once, reused across both m-tiles (4 indep. chains)
      v16h bw = load_b(w2B, 64, kt, tt, lane);
      v16h bp = load_b(protosB, 64, kt, tt, lane);
      lg[0] = wmma_f16(load_a(hh, 0, kt, ln, hi), bw, lg[0]);
      lg[1] = wmma_f16(load_a(hh, 1, kt, ln, hi), bw, lg[1]);
      sm[0] = wmma_f16(load_a(tokh, 0, kt, ln, hi), bp, sm[0]);
      sm[1] = wmma_f16(load_a(tokh, 1, kt, ln, hi), bp, sm[1]);
    }
    const float bias = mlp_b2[tt * 16 + ln];
#pragma unroll
    for (int mt = 0; mt < 2; ++mt) {
#pragma unroll
      for (int v = 0; v < 8; ++v) {
        const float lval = lg[mt][v] + bias;
        const float sval = sm[mt][v];
        float tmax = lval;
        for (int off = 1; off < 16; off <<= 1)
          tmax = fmaxf(tmax, __shfl_xor(tmax, off, 32));
        const float nm = fmaxf(rmax[mt][v], tmax);
        const float e = __expf(lval - nm);
        float se = e, sn = e * sval;
        for (int off = 1; off < 16; off <<= 1) {
          se += __shfl_xor(se, off, 32);
          sn += __shfl_xor(sn, off, 32);
        }
        const float scale = __expf(rmax[mt][v] - nm);
        rden[mt][v] = rden[mt][v] * scale + se;
        rnum[mt][v] = rnum[mt][v] * scale + sn;
        rmax[mt][v] = nm;
      }
    }
  }

  // ---- per-wave partials -> LDS, then combine across waves ----
  if (ln == 0) {
#pragma unroll
    for (int mt = 0; mt < 2; ++mt)
#pragma unroll
      for (int v = 0; v < 8; ++v) {
        const int m = mt * 16 + v + 8 * hi;
        parts[w][m][0] = rmax[mt][v];
        parts[w][m][1] = rden[mt][v];
        parts[w][m][2] = rnum[mt][v];
      }
  }
  __syncthreads();
  if (tid < MT) {
    const int m = tid;
    float gm = -1e30f;
    for (int q = 0; q < 8; ++q) gm = fmaxf(gm, parts[q][m][0]);
    float den = 0.0f, num = 0.0f;
    for (int q = 0; q < 8; ++q) {
      const float sc = __expf(parts[q][m][0] - gm);
      den += parts[q][m][1] * sc;
      num += parts[q][m][2] * sc;
    }
    const float sim = num / den;
    out[g0 + m] = fminf(fmaxf(sim * 0.5f + 0.5f, 0.0f), 1.0f);
  }
}

// ---------------- launch ----------------

extern "C" void kernel_launch(void* const* d_in, const int* in_sizes, int n_in,
                              void* d_out, int out_size, void* d_ws, size_t ws_size,
                              hipStream_t stream) {
  const float* x      = (const float*)d_in[0];
  const float* conv_w = (const float*)d_in[1];
  const float* conv_b = (const float*)d_in[2];
  const float* book   = (const float*)d_in[3];
  const float* mlp_w1 = (const float*)d_in[4];
  const float* mlp_b1 = (const float*)d_in[5];
  const float* mlp_w2 = (const float*)d_in[6];
  const float* mlp_b2 = (const float*)d_in[7];

  char* ws = (char*)d_ws;
  float* inv    = (float*)ws;                     // 1024 f32
  f16* convB    = (f16*)(ws + 4096);              // 65536 halfs
  f16* w1B      = convB + 65536;                  // 65536 halfs
  f16* w2B      = w1B + 65536;                    // 262144 halfs
  f16* protosB  = w2B + 262144;                   // 262144 halfs

  norm_kernel<<<TT, 256, 0, stream>>>(book, inv);
  // conv_w is [O,C] -> B[k=c][n=o] reads src[n*K+k]
  pack_kernel<<<(CCH * CCH) / 256, 256, 0, stream>>>(convB, conv_w, CCH, CCH, 1, nullptr);
  pack_kernel<<<(CCH * HID) / 256, 256, 0, stream>>>(w1B, mlp_w1, CCH, HID, 0, nullptr);
  pack_kernel<<<(HID * TT) / 256, 256, 0, stream>>>(w2B, mlp_w2, HID, TT, 0, nullptr);
  // protos: B[k=c][n=t] = book[t][c] * inv[t]
  pack_kernel<<<(CCH * TT) / 256, 256, 0, stream>>>(protosB, book, CCH, TT, 1, inv);

  const int total_tokens = 2 * NTOK;
  tokenbook_fused<<<total_tokens / MT, 256, 0, stream>>>(
      x, conv_b, mlp_b1, mlp_b2, convB, w1B, w2B, protosB, (float*)d_out);
}
